// DiffAttn_83159156785446
// MI455X (gfx1250) — matrix operand order repaired and checked by reading, not compiled, not run on hardware
//
#include <hip/hip_runtime.h>
#include <hip/hip_bf16.h>

typedef __attribute__((ext_vector_type(16))) __bf16 v16bf;
typedef __attribute__((ext_vector_type(8)))  float  v8f;

// ---------------------------------------------------------------------------
// WMMA helper: D = A(16x32 bf16) * B(32x16 bf16) + C(16x16 f32)
// ---------------------------------------------------------------------------
__device__ __forceinline__ v8f wmma_bf16(v16bf a, v16bf b, v8f c) {
    return __builtin_amdgcn_wmma_f32_16x16x32_bf16(
        /*neg_a=*/false, a, /*neg_b=*/false, b,
        /*c_mod=*/(short)0, c, /*reuse_a=*/false, /*reuse_b=*/false);
}

// ---------------------------------------------------------------------------
// CDNA5 async memory->LDS copy (ASYNCcnt-tracked, no VGPR round-trip).
// Generic LDS pointers truncate to the LDS byte offset in addr[31:0].
// ---------------------------------------------------------------------------
__device__ __forceinline__ void async_load_b128(void* lds_dst, const void* gsrc) {
    unsigned           ldsa = (unsigned)(size_t)lds_dst;
    unsigned long long ga   = (unsigned long long)(size_t)gsrc;
    asm volatile("global_load_async_to_lds_b128 %0, %1, off"
                 :: "v"(ldsa), "v"(ga) : "memory");
}
__device__ __forceinline__ void wait_async0() {
    asm volatile("s_wait_asynccnt 0x0" ::: "memory");
}

// A-matrix 16x32 bf16 fragment (ISA 7.12.2): lane L -> row m0+(L&15),
// halves [k .. k+7] and [k+16 .. k+23] with k = k0 + (L>>4)*8.
__device__ __forceinline__ v16bf load_frag_a(const __bf16* base, int pitch,
                                             int lane, int m0, int k0) {
    int m = m0 + (lane & 15);
    int k = k0 + ((lane >> 4) << 3);
    const __bf16* p = base + (size_t)m * pitch + k;
    union { uint4 u[2]; v16bf v; } t;
    t.u[0] = *(const uint4*)(p);
    t.u[1] = *(const uint4*)(p + 16);
    return t.v;
}

// B-matrix 32x16 bf16 fragment: lane L -> col n0+(L&15),
// halves [k .. k+7] and [k+16 .. k+23] with k = k0 + (L>>4)*8; base stored [n][k].
__device__ __forceinline__ v16bf load_frag_b(const __bf16* base, int pitch,
                                             int lane, int n0, int k0) {
    int n = n0 + (lane & 15);
    int k = k0 + ((lane >> 4) << 3);
    const __bf16* p = base + (size_t)n * pitch + k;
    union { uint4 u[2]; v16bf v; } t;
    t.u[0] = *(const uint4*)(p);
    t.u[1] = *(const uint4*)(p + 16);
    return t.v;
}

// ---------------------------------------------------------------------------
// fp32 -> bf16 conversion (vectorized x4)
// ---------------------------------------------------------------------------
__global__ void cvt_bf16(const float* __restrict__ in, __bf16* __restrict__ out,
                         int n4) {
    int i = blockIdx.x * blockDim.x + threadIdx.x;
    if (i < n4) {
        float4 f = ((const float4*)in)[i];
        __bf16 h[4] = {(__bf16)f.x, (__bf16)f.y, (__bf16)f.z, (__bf16)f.w};
        *(uint2*)(out + (size_t)i * 4) = *(const uint2*)h;
    }
}

// ---------------------------------------------------------------------------
// Projection GEMM: Out[m][n] = sum_k A[m][k] * W[n][k] + bias[n]   (NT GEMM)
// 256 threads = 8 waves; 64x64 tile; BK=64; wave -> 16x32 subtile.
// Double-buffered: async-stage chunk k+1 while WMMAs consume chunk k.
// ---------------------------------------------------------------------------
__global__ void __launch_bounds__(256)
proj_gemm(const __bf16* __restrict__ A, const __bf16* __restrict__ W,
          const float* __restrict__ bias, __bf16* __restrict__ Out,
          int M, int N, int K) {
    constexpr int P    = 80;               // LDS pitch in halves (160B rows)
    constexpr int TILE = 64 * P * 2;       // bytes per tile buffer
    extern __shared__ __align__(16) char smem[];
    __bf16* As0 = (__bf16*)(smem);
    __bf16* As1 = (__bf16*)(smem + TILE);
    __bf16* Ws0 = (__bf16*)(smem + 2 * TILE);
    __bf16* Ws1 = (__bf16*)(smem + 3 * TILE);

    const int tid  = threadIdx.x;
    const int wave = tid >> 5, lane = tid & 31;
    const int m0 = blockIdx.x * 64, n0 = blockIdx.y * 64;
    const int wm = (wave & 3) * 16;        // wave's M offset (0..48)
    const int wn = (wave >> 2) * 32;       // wave's N offset (0 or 32)

    auto stage = [&](__bf16* Ad, __bf16* Wd, int k0) {
#pragma unroll
        for (int i = 0; i < 2; ++i) {
            int flat = tid + i * 256;              // 512 uint4 slots per tile
            int r = flat >> 3, c8 = (flat & 7) << 3;
            async_load_b128(Ad + r * P + c8,
                            A + (size_t)(m0 + r) * K + k0 + c8);
            async_load_b128(Wd + r * P + c8,
                            W + (size_t)(n0 + r) * K + k0 + c8);
        }
    };

    v8f c0 = {}, c1 = {};

    stage(As0, Ws0, 0);
    wait_async0();
    __syncthreads();

    int cur = 0;
    for (int k0 = 0; k0 < K; k0 += 64) {
        __bf16* Ac = cur ? As1 : As0;
        __bf16* Wc = cur ? Ws1 : Ws0;
        if (k0 + 64 < K) stage(cur ? As0 : As1, cur ? Ws0 : Ws1, k0 + 64);

        // hoist all fragments of this BK chunk, then fire 4 WMMAs
        v16bf af0 = load_frag_a(Ac, P, lane, wm, 0);
        v16bf af1 = load_frag_a(Ac, P, lane, wm, 32);
        v16bf b00 = load_frag_b(Wc, P, lane, wn,      0);
        v16bf b01 = load_frag_b(Wc, P, lane, wn + 16, 0);
        v16bf b10 = load_frag_b(Wc, P, lane, wn,      32);
        v16bf b11 = load_frag_b(Wc, P, lane, wn + 16, 32);
        c0 = wmma_bf16(af0, b00, c0);
        c1 = wmma_bf16(af0, b01, c1);
        c0 = wmma_bf16(af1, b10, c0);
        c1 = wmma_bf16(af1, b11, c1);

        wait_async0();
        __syncthreads();
        cur ^= 1;
    }

    // Epilogue: C layout -> VGPR i: rows wm+i (lanes 0-15), wm+i+8 (lanes 16-31)
    const int rbase = wm + ((lane >> 4) << 3);
    const int nc    = wn + (lane & 15);
#pragma unroll
    for (int i = 0; i < 8; ++i) {
        int m = m0 + rbase + i;
        float v0 = c0[i] + bias[n0 + nc];
        float v1 = c1[i] + bias[n0 + nc + 16];
        Out[(size_t)m * N + n0 + nc]      = (__bf16)v0;
        Out[(size_t)m * N + n0 + nc + 16] = (__bf16)v1;
    }
}

// ---------------------------------------------------------------------------
// Differential flash attention, two online softmaxes.
// Grid: (S/32, B); 256 threads = 8 waves. Br=32 queries, Bc=64 keys.
// ---------------------------------------------------------------------------
__global__ void __launch_bounds__(256)
diff_attn(const __bf16* __restrict__ Qb, const __bf16* __restrict__ Kb,
          const __bf16* __restrict__ Vb, const float* __restrict__ lam,
          float* __restrict__ Out) {
    constexpr int S = 2048, D = 512, E2 = 1024;
    constexpr int PQ = 1040, PK = 272, PV = 80, PS = 68, PP = 80;
    constexpr float SCALE = 0.04419417382415922f;  // 1/sqrt(512)
    const float NEG_INF = -__builtin_inff();

    extern __shared__ __align__(16) char smem[];
    __bf16* Qs  = (__bf16*)(smem);                       // 32 x PQ
    __bf16* Ksm = (__bf16*)(smem + 66560);               // 64 x PK (both heads)
    __bf16* Vt  = (__bf16*)(smem + 101376);              // 512 x PV (d-major)
    float*  S1  = (float*) (smem + 183296);              // 32 x PS
    float*  S2  = (float*) (smem + 192000);
    __bf16* P1  = (__bf16*)(smem + 200704);              // 32 x PP
    __bf16* P2  = (__bf16*)(smem + 205824);
    float*  m1  = (float*) (smem + 210944);
    float*  l1  = m1 + 32;
    float*  cc1 = m1 + 64;
    float*  m2  = m1 + 96;
    float*  l2  = m1 + 128;
    float*  cc2 = m1 + 160;

    const int tid = threadIdx.x, wave = tid >> 5, lane = tid & 31;
    const int qb = blockIdx.x, b = blockIdx.y;

    const __bf16* Qg = Qb + ((size_t)b * S + (size_t)qb * 32) * E2;
    const __bf16* Kg = Kb + (size_t)b * S * E2;
    const __bf16* Vg = Vb + (size_t)b * S * D;

    // Q tile -> LDS (32 x 1024 halves) via async copy
#pragma unroll
    for (int i = 0; i < 16; ++i) {
        int flat = tid + i * 256;              // 4096 uint4 slots
        int r = flat >> 7, c8 = (flat & 127) << 3;
        async_load_b128(Qs + r * PQ + c8, Qg + (size_t)r * E2 + c8);
    }
    if (tid < 32) { m1[tid] = NEG_INF; l1[tid] = 0.f; m2[tid] = NEG_INF; l2[tid] = 0.f; }

    v8f acc1[2][4] = {};
    v8f acc2[2][4] = {};
    wait_async0();
    __syncthreads();

    const int qmax = qb * 32 + 31;
    const int nkb  = qmax / 64 + 1;            // causal: skip fully-masked blocks

    for (int kb = 0; kb < nkb; ++kb) {
        // ---------------- scores: wave = (head h, 16-key slice nf) ----------
        const int h  = wave & 1;
        const int nf = wave >> 1;
        v8f s0 = {}, s1 = {};

        for (int ch = 0; ch < 4; ++ch) {       // 128 K-cols per head per chunk
            // async-stage K chunk: 64 rows x (128 head0 | 128 head1) halves
#pragma unroll
            for (int i = 0; i < 8; ++i) {
                int flat = tid + i * 256;
                int r = flat >> 5, c8 = (flat & 31) << 3;
                int gcol = (c8 < 128) ? (ch * 128 + c8) : (512 + ch * 128 + (c8 - 128));
                async_load_b128(Ksm + r * PK + c8,
                                Kg + (size_t)(kb * 64 + r) * E2 + gcol);
            }
            wait_async0();
            __syncthreads();
#pragma unroll
            for (int ks = 0; ks < 128; ks += 32) {
                int kq = h * 512 + ch * 128 + ks;        // column in Q tile
                v16bf a0 = load_frag_a(Qs, PQ, lane, 0, kq);
                v16bf a1 = load_frag_a(Qs, PQ, lane, 16, kq);
                v16bf bf = load_frag_b(Ksm, PK, lane, nf * 16, h * 128 + ks);
                s0 = wmma_bf16(a0, bf, s0);
                s1 = wmma_bf16(a1, bf, s1);
            }
            __syncthreads();
        }

        // store raw scores to LDS
        float* Sh = h ? S2 : S1;
#pragma unroll
        for (int i = 0; i < 8; ++i) {
            int rb = i + ((lane >> 4) << 3);
            Sh[(rb)      * PS + nf * 16 + (lane & 15)] = s0[i];
            Sh[(rb + 16) * PS + nf * 16 + (lane & 15)] = s1[i];
        }

        // V block -> LDS transposed (Vt[d][key]) while scores settle
#pragma unroll
        for (int i = 0; i < 16; ++i) {
            int flat = tid + i * 256;          // 4096 uint4 slots
            int key = flat >> 6, d0 = (flat & 63) << 3;
            uint4 u = *(const uint4*)(Vg + (size_t)(kb * 64 + key) * D + d0);
            const __bf16* hp = (const __bf16*)&u;
#pragma unroll
            for (int j = 0; j < 8; ++j) Vt[(d0 + j) * PV + key] = hp[j];
        }
        __syncthreads();

        // ---------------- online softmax (64 threads: 32 rows x 2 heads) ----
        if (tid < 64) {
            const int r = tid & 31, hh = tid >> 5;
            float* Ss  = hh ? S2 : S1;
            __bf16* Pp = hh ? P2 : P1;
            float* mm  = hh ? m2 : m1;
            float* ll  = hh ? l2 : l1;
            float* ccf = hh ? cc2 : cc1;
            const int q = qb * 32 + r;

            float mold = mm[r];
            float bmax = NEG_INF;
            for (int cj = 0; cj < 64; ++cj) {
                int key = kb * 64 + cj;
                float v = (key <= q) ? Ss[r * PS + cj] * SCALE : NEG_INF;
                bmax = fmaxf(bmax, v);
            }
            float mnew = fmaxf(mold, bmax);
            float corr = (mold == NEG_INF) ? 0.0f : __expf(mold - mnew);
            if (mnew == NEG_INF) corr = 1.0f;

            float lsum = 0.f;
            for (int cj = 0; cj < 64; ++cj) {
                int key = kb * 64 + cj;
                float v = (key <= q) ? Ss[r * PS + cj] * SCALE : NEG_INF;
                float p = (mnew == NEG_INF) ? 0.0f : __expf(v - mnew);
                Pp[r * PP + cj] = (__bf16)p;
                lsum += p;
            }
            mm[r] = mnew;
            ll[r] = ll[r] * corr + lsum;
            ccf[r] = corr;
        }
        __syncthreads();

        // ---------------- PV: wave owns d-slice [wave*64, +64) ---------------
#pragma unroll
        for (int fm = 0; fm < 2; ++fm)
#pragma unroll
            for (int i = 0; i < 8; ++i) {
                int row = fm * 16 + i + ((lane >> 4) << 3);
                float k1 = cc1[row], k2 = cc2[row];
#pragma unroll
                for (int fn = 0; fn < 4; ++fn) {
                    acc1[fm][fn][i] *= k1;
                    acc2[fm][fn][i] *= k2;
                }
            }

#pragma unroll
        for (int ks = 0; ks < 64; ks += 32) {
            v16bf bv[4];
#pragma unroll
            for (int fn = 0; fn < 4; ++fn)
                bv[fn] = load_frag_b(Vt, PV, lane, wave * 64 + fn * 16, ks);
#pragma unroll
            for (int fm = 0; fm < 2; ++fm) {
                v16bf a1f = load_frag_a(P1, PP, lane, fm * 16, ks);
                v16bf a2f = load_frag_a(P2, PP, lane, fm * 16, ks);
#pragma unroll
                for (int fn = 0; fn < 4; ++fn) {
                    acc1[fm][fn] = wmma_bf16(a1f, bv[fn], acc1[fm][fn]);
                    acc2[fm][fn] = wmma_bf16(a2f, bv[fn], acc2[fm][fn]);
                }
            }
        }
        __syncthreads();
    }

    // ---------------- combine: out = acc1/l1 - lambda * acc2/l2 -------------
    const float lambda = __expf(lam[0]) + 0.05f;
#pragma unroll
    for (int fm = 0; fm < 2; ++fm)
#pragma unroll
        for (int i = 0; i < 8; ++i) {
            int row = fm * 16 + i + ((lane >> 4) << 3);
            float il1 = l1[row] > 0.f ? 1.0f / l1[row] : 0.f;
            float il2 = l2[row] > 0.f ? 1.0f / l2[row] : 0.f;
            int q = qb * 32 + row;
#pragma unroll
            for (int fn = 0; fn < 4; ++fn) {
                int d = wave * 64 + fn * 16 + (lane & 15);
                Out[((size_t)b * S + q) * D + d] =
                    acc1[fm][fn][i] * il1 - lambda * acc2[fm][fn][i] * il2;
            }
        }
}

// ---------------------------------------------------------------------------
extern "C" void kernel_launch(void* const* d_in, const int* in_sizes, int n_in,
                              void* d_out, int out_size, void* d_ws, size_t ws_size,
                              hipStream_t stream) {
    constexpr int B = 4, S = 2048, E = 1024, D = 512;
    constexpr size_t M = (size_t)B * S;

    const float* X   = (const float*)d_in[0];
    /* d_in[1] = mask (bool) — causal mask recomputed on device */
    const float* Wq  = (const float*)d_in[2];
    const float* bq  = (const float*)d_in[3];
    const float* Wk  = (const float*)d_in[4];
    const float* bk  = (const float*)d_in[5];
    const float* Wv  = (const float*)d_in[6];
    const float* bv  = (const float*)d_in[7];
    const float* lam = (const float*)d_in[8];
    float* Out = (float*)d_out;

    char* ws = (char*)d_ws;
    __bf16* Xb  = (__bf16*)ws; ws += M * E * 2;
    __bf16* Wqb = (__bf16*)ws; ws += (size_t)(2 * D) * E * 2;
    __bf16* Wkb = (__bf16*)ws; ws += (size_t)(2 * D) * E * 2;
    __bf16* Wvb = (__bf16*)ws; ws += (size_t)D * E * 2;
    __bf16* Qd  = (__bf16*)ws; ws += M * (2 * D) * 2;
    __bf16* Kd  = (__bf16*)ws; ws += M * (2 * D) * 2;
    __bf16* Vd  = (__bf16*)ws; ws += M * D * 2;

    // fp32 -> bf16
    auto cvt = [&](const float* src, __bf16* dst, size_t n) {
        int n4 = (int)(n / 4);
        cvt_bf16<<<(n4 + 255) / 256, 256, 0, stream>>>(src, dst, n4);
    };
    cvt(X,  Xb,  M * E);
    cvt(Wq, Wqb, (size_t)(2 * D) * E);
    cvt(Wk, Wkb, (size_t)(2 * D) * E);
    cvt(Wv, Wvb, (size_t)D * E);

    // Q/K/V projections (NT GEMM with WMMA bf16, double-buffered async LDS)
    constexpr int SMEM_PROJ = 4 * 64 * 80 * 2;  // 40 KB (2 x double buffer)
    proj_gemm<<<dim3(M / 64, (2 * D) / 64), 256, SMEM_PROJ, stream>>>(
        Xb, Wqb, bq, Qd, (int)M, 2 * D, E);
    proj_gemm<<<dim3(M / 64, (2 * D) / 64), 256, SMEM_PROJ, stream>>>(
        Xb, Wkb, bk, Kd, (int)M, 2 * D, E);
    proj_gemm<<<dim3(M / 64, D / 64), 256, SMEM_PROJ, stream>>>(
        Xb, Wvb, bv, Vd, (int)M, D, E);

    // Differential flash attention
    constexpr int SMEM_ATTN = 211712;  // ~207 KB dynamic LDS (WGP has 320 KB)
    hipFuncSetAttribute((const void*)diff_attn,
                        hipFuncAttributeMaxDynamicSharedMemorySize, SMEM_ATTN);
    diff_attn<<<dim3(S / 32, B), 256, SMEM_ATTN, stream>>>(Qd, Kd, Vd, lam, Out);
}
